// MySGNNMD_1778116460983
// MI455X (gfx1250) — matrix operand order repaired and checked by reading, not compiled, not run on hardware
//
#include <hip/hip_runtime.h>
#include <hip/hip_bf16.h>
#include <math.h>

typedef __attribute__((ext_vector_type(16))) _Float16 v16h;
typedef __attribute__((ext_vector_type(8)))  _Float16 v8h;
typedef __attribute__((ext_vector_type(8)))  float    v8f;

#define NNODES 512
#define DPOS 16
#define DNEG 16
#define BIO 64
#define TOPO 16
#define KSEL 25
#define HID 8
#define MROWS 64                 // 2*(DP+DN)
#define FEAT (KSEL*(TOPO+BIO))   // 2000
#define FEAT_PAD 2016            // 63 * 32
#define KSTEPS 63
#define EPOS 8192
#define ETOT 16384

// ---------------- ws layout (bytes) ----------------
#define FH_OFF   0        // half[2][512][64]   131072
#define BP_OFF   131072   // half[63][32][16]   64512
#define TH_OFF   196608   // half[2][64][16]    4096
#define FR_OFF   200704   // short[2][2016]     8064
#define FC_OFF   208896   // short[2][2016]     8064
#define NB_OFF   217088   // short[4][512][16]  65536

// ---- 1. fp32 -> fp16 feature tables (gene | pheo concatenated) ----
__global__ void convert_feats_kernel(const float* __restrict__ gf,
                                     const float* __restrict__ pf,
                                     _Float16* __restrict__ fh) {
  int tid = blockIdx.x * blockDim.x + threadIdx.x;
  if (tid < NNODES * BIO)           fh[tid] = (_Float16)gf[tid];
  else if (tid < 2 * NNODES * BIO)  fh[tid] = (_Float16)pf[tid - NNODES * BIO];
}

// ---- 2. pack mw1 (2000x8) into WMMA-B fragments: bp[t][lane][h] ----
// B (32x16): lane%16 = N column, lane/16 = g, half h -> K = g*8 + h (h<8), 16 + g*8 + (h-8) (h>=8)
__global__ void pack_mw1_kernel(const float* __restrict__ mw1,
                                _Float16* __restrict__ bp) {
  int tid = blockIdx.x * blockDim.x + threadIdx.x;
  if (tid >= KSTEPS * 512) return;
  int t    = tid >> 9;
  int rem  = tid & 511;
  int lane = rem >> 4;
  int h    = rem & 15;
  int n    = lane & 15;
  int gsel = lane >> 4;
  int k    = t * 32 + ((h < 8) ? (gsel * 8 + h) : (16 + gsel * 8 + (h - 8)));
  float v  = (k < FEAT && n < HID) ? mw1[k * HID + n] : 0.0f;
  bp[tid] = (_Float16)v;
}

// ---- 3. stable neighbor tables (matches stable argsort order) ----
__global__ void build_nb_kernel(const int* __restrict__ pe,
                                const int* __restrict__ ne,
                                short* __restrict__ nb) {
  int tid = blockIdx.x * blockDim.x + threadIdx.x;
  if (tid >= 4 * NNODES) return;
  int tbl  = tid >> 9;
  int node = tid & (NNODES - 1);
  const int* key; const int* val;
  if      (tbl == 0) { key = pe + EPOS; val = pe; }
  else if (tbl == 1) { key = ne + EPOS; val = ne; }
  else if (tbl == 2) { key = pe;        val = pe + EPOS; }
  else               { key = ne;        val = ne + EPOS; }
  int cnt = 0;
  for (int e = 0; e < EPOS && cnt < 16; ++e)
    if (key[e] == node) nb[(tbl * NNODES + node) * 16 + (cnt++)] = (short)val[e];
}

// ---- 4. tiny GCN + SortPool top-k; blockIdx.x == state (0 or 1) ----
__global__ __launch_bounds__(128) void topo_kernel(
    const float* __restrict__ tw,  const float* __restrict__ tb,
    const float* __restrict__ pw1, const float* __restrict__ pb1,
    const float* __restrict__ pw2, const float* __restrict__ pb2,
    const float* __restrict__ nw1, const float* __restrict__ nb1,
    const float* __restrict__ nw2, const float* __restrict__ nb2,
    _Float16* __restrict__ topoh, short* __restrict__ frow, short* __restrict__ fcol) {
  const int st = blockIdx.x;          // state
  const int pg = DPOS - st;
  const int ng = DNEG + st - 1;
  const int pp = pg, nq = ng;         // gl = 1+pg+ng = 32 always
  __shared__ float H[65 * 16];
  __shared__ float X[65 * 16];
  __shared__ float TP[65 * 16];
  __shared__ float SV[64];
  __shared__ int   TOPI[KSEL];

  for (int i = threadIdx.x; i < 65 * 16; i += blockDim.x) TP[i] = 0.0f;
  __syncthreads();

  for (int graph = 0; graph < 2; ++graph) {   // 0 = pos graph, 1 = neg graph
    const float* W1 = graph == 0 ? pw1 : nw1;
    const float* B1 = graph == 0 ? pb1 : nb1;
    const float* W2 = graph == 0 ? pw2 : nw2;
    const float* B2 = graph == 0 ? pb2 : nb2;
    for (int i = threadIdx.x; i < 65 * 16; i += blockDim.x) {
      int v = i >> 4, c = i & 15;
      int lab;
      if (v <= 1)                lab = 0;
      else if (v < 2 + pg)       lab = 2;
      else if (v < 2 + pg + ng)  lab = 4;
      else if (v == 33)          lab = 1;
      else if (v < 34 + pp)      lab = 3;
      else                       lab = 5;
      H[i] = W1[lab * 16 + c];
    }
    __syncthreads();

    for (int layer = 0; layer < 2; ++layer) {
      if (layer == 1) {                        // H = X @ W2
        for (int i = threadIdx.x; i < 65 * 16; i += blockDim.x) {
          int v = i >> 4, c = i & 15;
          float s = 0.0f;
          for (int k = 0; k < 16; ++k) s += X[v * 16 + k] * W2[k * 16 + c];
          H[i] = s;
        }
        __syncthreads();
      }
      const float* Bv = (layer == 0) ? B1 : B2;
      for (int i = threadIdx.x; i < 65 * 16; i += blockDim.x) {
        int v = i >> 4, c = i & 15;
        float acc;
        if (graph == 0) {
          float dv = (v == 33) ? (float)(pg + 1)
                               : ((v >= 34 && v < 34 + pp) ? 2.0f : 1.0f);
          float rv = rsqrtf(dv);
          acc = H[i] * rv * rv;
          if (v == 33) { for (int s2 = 2; s2 < 2 + pg; ++s2) acc += H[s2 * 16 + c] * rv; }
          else if (v >= 34 && v < 34 + pp) acc += H[16 + c] * rv;
        } else {
          float dv = (v == 33) ? (float)(ng + 1)
                               : ((v >= 34 + pp && v < 34 + pp + nq) ? 2.0f : 1.0f);
          float rv = rsqrtf(dv);
          acc = H[i] * rv * rv;
          if (v == 33) { for (int s2 = 2 + pg; s2 < 2 + pg + ng; ++s2) acc += H[s2 * 16 + c] * rv; }
          else if (v >= 34 + pp && v < 34 + pp + nq) acc += H[16 + c] * rv;
        }
        X[i] = tanhf(acc + Bv[c]);
      }
      __syncthreads();
    }
    for (int i = threadIdx.x; i < 65 * 16; i += blockDim.x) TP[i] += X[i];
    __syncthreads();
  }

  for (int r = threadIdx.x; r < MROWS; r += blockDim.x) {
    float s = tb[0];
    for (int c = 0; c < 16; ++c) s += TP[(r + 1) * 16 + c] * tw[c];
    SV[r] = s;
  }
  __syncthreads();
  for (int r = threadIdx.x; r < MROWS; r += blockDim.x) {
    float sr = SV[r];
    int rank = 0;
    for (int j = 0; j < MROWS; ++j) {
      float sj = SV[j];
      if (sj > sr || (sj == sr && j < r)) ++rank;
    }
    if (rank < KSEL) TOPI[rank] = r;
  }
  __syncthreads();
  for (int i = threadIdx.x; i < MROWS * 16; i += blockDim.x)
    topoh[st * 1024 + i] = (_Float16)TP[i + 16];
  for (int f = threadIdx.x; f < FEAT_PAD; f += blockDim.x) {
    if (f < FEAT) { frow[st * FEAT_PAD + f] = (short)TOPI[f / 80];
                    fcol[st * FEAT_PAD + f] = (short)(f % 80); }
    else          { frow[st * FEAT_PAD + f] = 0; fcol[st * FEAT_PAD + f] = 0; }
  }
}

// ---- 5. edge scoring: async-gather A tile to LDS, then branch-free WMMA ----
__global__ __launch_bounds__(32) void edge_score_kernel(
    const int* __restrict__ pe, const int* __restrict__ ne,
    const _Float16* __restrict__ FH, const _Float16* __restrict__ BP,
    const _Float16* __restrict__ TH, const short* __restrict__ FRg,
    const short* __restrict__ NBg,
    const float* __restrict__ mb1, const float* __restrict__ mw2,
    const float* __restrict__ mb2, float* __restrict__ out) {
  __shared__ __align__(16) _Float16 AT[16 * FEAT_PAD];   // 64512 B A-tile
  __shared__ int      OFF[16 * MROWS];
  __shared__ _Float16 LT[MROWS * TOPO];
  __shared__ short    ROWS[KSEL];
  __shared__ float    ACC[16 * 16];

  const int base = blockIdx.x * 16;
  const int st   = (base < EPOS) ? 1 : 0;
  const int tid  = threadIdx.x;

  for (int idx = tid; idx < 16 * MROWS; idx += 32) {
    int mm = idx >> 6, r = idx & 63;
    int e = base + mm;
    int g, p;
    if (st == 1) { g = pe[e];        p = pe[EPOS + e]; }
    else         { g = ne[e - EPOS]; p = ne[e]; }      // ne[EPOS + (e-EPOS)]
    int node;
    if      (r < 16) node = NBg[(0 * NNODES + p) * 16 + r];
    else if (r < 32) node = NBg[(1 * NNODES + p) * 16 + (r - 16)];
    else if (r < 48) node = NBg[(2 * NNODES + g) * 16 + (r - 32)];
    else             node = NBg[(3 * NNODES + g) * 16 + (r - 48)];
    OFF[idx] = ((r < 32) ? 0 : NNODES * BIO) + node * BIO;
  }
  for (int i = tid; i < MROWS * TOPO; i += 32) LT[i] = TH[st * 1024 + i];
  if (tid < KSEL) ROWS[tid] = FRg[st * FEAT_PAD + tid * 80];
  __syncthreads();

  // topo columns of each selected row (broadcast over edges)
  for (int idx = tid; idx < 16 * KSEL * TOPO; idx += 32) {
    int mm  = idx / (KSEL * TOPO);
    int rem = idx - mm * (KSEL * TOPO);
    int s = rem >> 4, c = rem & 15;
    AT[mm * FEAT_PAD + s * 80 + c] = LT[ROWS[s] * TOPO + c];
  }
  // zero pad features 2000..2015
  for (int idx = tid; idx < 16 * 16; idx += 32) {
    int mm = idx >> 4, c = idx & 15;
    AT[mm * FEAT_PAD + FEAT + c] = (_Float16)0.0f;
  }
  // async gather of bio rows: 16 edges * 25 rows * 128B, 16B per lane per op
  for (int q = tid; q < 16 * KSEL * 8; q += 32) {
    int mm  = q / (KSEL * 8);
    int rem = q - mm * (KSEL * 8);
    int s = rem >> 3, ch = rem & 7;
    const _Float16* gp = FH + OFF[mm * MROWS + ROWS[s]] + ch * 8;
    unsigned lds = (unsigned)(unsigned long long)(uintptr_t)
                   &AT[mm * FEAT_PAD + s * 80 + TOPO + ch * 8];
    asm volatile("global_load_async_to_lds_b128 %0, %1, off"
                 :: "v"(lds), "v"(gp) : "memory");
  }
  asm volatile("s_wait_asynccnt 0x0" ::: "memory");
  __syncthreads();

  // branch-free WMMA loop: 2x ds_load_b128 (A) + B + wmma per K-step
  const int m    = tid & 15;   // A-matrix M row
  const int gsel = tid >> 4;   // K-half group
  const _Float16* am = &AT[m * FEAT_PAD];
  v8f acc = {};
  for (int t = 0; t < KSTEPS; ++t) {
    v8h lo = *(const v8h*)(am + t * 32 + gsel * 8);         // K = t*32 + g*8 + h
    v8h hi = *(const v8h*)(am + t * 32 + 16 + gsel * 8);    // K = t*32 + 16 + g*8 + h
    v16h a = __builtin_shufflevector(lo, hi, 0, 1, 2, 3, 4, 5, 6, 7,
                                             8, 9, 10, 11, 12, 13, 14, 15);
    v16h b = *(const v16h*)(BP + (size_t)t * 512 + tid * 16);
    acc = __builtin_amdgcn_wmma_f32_16x16x32_f16(false, a, false, b,
                                                 (short)0, acc, false, false);
  }

  // C/D layout: lanes 0-15 -> M=r, lanes 16-31 -> M=r+8; N = lane&15
#pragma unroll
  for (int r = 0; r < 8; ++r) ACC[(r + gsel * 8) * 16 + m] = acc[r];
  __syncthreads();

  if (tid < 16) {
    float s = mb2[0];
    for (int n = 0; n < HID; ++n) {
      float hv = ACC[tid * 16 + n] + mb1[n];
      hv = fmaxf(hv, 0.0f);
      s += hv * mw2[n];
    }
    out[1 + base + tid]        = s;
    out[1 + ETOT + base + tid] = (st == 1) ? 1.0f : 0.0f;
  }
}

// ---- 6. deterministic loss reduction ----
__global__ __launch_bounds__(256) void loss_kernel(float* __restrict__ out) {
  __shared__ float sA[256], sB[256], sP[256], sN[256];
  float A = 0.0f, B = 0.0f, P = 0.0f, Nc = 0.0f;
  for (int e = threadIdx.x; e < ETOT; e += 256) {
    float s = out[1 + e];
    float y = out[1 + ETOT + e];
    float l1p = log1pf(expf(-fabsf(s)));
    float spn = fmaxf(-s, 0.0f) + l1p;    // softplus(-s)
    float spp = fmaxf(s, 0.0f) + l1p;     // softplus(s)
    A  += y * spn;
    B  += (1.0f - y) * spp;
    P  += y;
    Nc += (1.0f - y);
  }
  sA[threadIdx.x] = A; sB[threadIdx.x] = B; sP[threadIdx.x] = P; sN[threadIdx.x] = Nc;
  __syncthreads();
  for (int off = 128; off > 0; off >>= 1) {
    if (threadIdx.x < off) {
      sA[threadIdx.x] += sA[threadIdx.x + off];
      sB[threadIdx.x] += sB[threadIdx.x + off];
      sP[threadIdx.x] += sP[threadIdx.x + off];
      sN[threadIdx.x] += sN[threadIdx.x + off];
    }
    __syncthreads();
  }
  if (threadIdx.x == 0) {
    float pw = sN[0] / sP[0];
    out[0] = (pw * sA[0] + sB[0]) / (float)ETOT;
  }
}

extern "C" void kernel_launch(void* const* d_in, const int* in_sizes, int n_in,
                              void* d_out, int out_size, void* d_ws, size_t ws_size,
                              hipStream_t stream) {
  const int*   pe  = (const int*)d_in[0];
  const int*   ne  = (const int*)d_in[1];
  const float* gf  = (const float*)d_in[2];
  const float* pf  = (const float*)d_in[3];
  const float* tw  = (const float*)d_in[4];
  const float* tb  = (const float*)d_in[5];
  const float* pw1 = (const float*)d_in[6];
  const float* pb1 = (const float*)d_in[7];
  const float* pw2 = (const float*)d_in[8];
  const float* pb2 = (const float*)d_in[9];
  const float* nw1 = (const float*)d_in[10];
  const float* nb1 = (const float*)d_in[11];
  const float* nw2 = (const float*)d_in[12];
  const float* nb2 = (const float*)d_in[13];
  const float* mw1 = (const float*)d_in[14];
  const float* mb1 = (const float*)d_in[15];
  const float* mw2 = (const float*)d_in[16];
  const float* mb2 = (const float*)d_in[17];

  char* ws = (char*)d_ws;
  _Float16* FH = (_Float16*)(ws + FH_OFF);
  _Float16* BP = (_Float16*)(ws + BP_OFF);
  _Float16* TH = (_Float16*)(ws + TH_OFF);
  short*    FR = (short*)(ws + FR_OFF);
  short*    FC = (short*)(ws + FC_OFF);
  short*    NB = (short*)(ws + NB_OFF);
  float*    out = (float*)d_out;

  convert_feats_kernel<<<(2 * NNODES * BIO + 255) / 256, 256, 0, stream>>>(gf, pf, FH);
  pack_mw1_kernel<<<(KSTEPS * 512 + 255) / 256, 256, 0, stream>>>(mw1, BP);
  build_nb_kernel<<<(4 * NNODES + 255) / 256, 256, 0, stream>>>(pe, ne, NB);
  topo_kernel<<<2, 128, 0, stream>>>(tw, tb, pw1, pb1, pw2, pb2,
                                     nw1, nb1, nw2, nb2, TH, FR, FC);
  edge_score_kernel<<<ETOT / 16, 32, 0, stream>>>(pe, ne, FH, BP, TH, FR, NB,
                                                  mb1, mw2, mb2, out);
  loss_kernel<<<1, 256, 0, stream>>>(out);
}
